// MultiHeadAttention_43894565765718
// MI455X (gfx1250) — matrix operand order repaired and checked by reading, not compiled, not run on hardware
//
#include <hip/hip_runtime.h>
#include <hip/hip_bf16.h>

typedef __attribute__((ext_vector_type(8)))  __bf16 v8bf;
typedef __attribute__((ext_vector_type(16))) __bf16 v16bf;
typedef __attribute__((ext_vector_type(8)))  float  v8f;

#define SEQ    4096
#define DMODEL 768
#define HEADS  12
#define HDIM   64
#define KT_ALL 24   /* 768/32 k-tiles */
#define NT_ALL 48   /* 768/16 n-tiles */

__device__ __forceinline__ v8f wmma_bf16(v16bf a, v16bf b, v8f c) {
    return __builtin_amdgcn_wmma_f32_16x16x32_bf16(false, a, false, b,
                                                   (short)0, c, false, false);
}

// A-fragment (16x32 bf16) from a row-major row pointer, contiguous K layout:
// lanes 0-15: row M=lane, K [0..15]; lanes 16-31: row M=lane-16,
// K [16..31].
__device__ __forceinline__ v16bf load_a_frag(const __bf16* row, int k0, int lane) {
    int sel = (lane >= 16) ? 16 : 0;
    v8bf lo = *(const v8bf*)(row + k0 + sel);
    v8bf hi = *(const v8bf*)(row + k0 + sel + 8);
    return __builtin_shufflevector(lo, hi, 0,1,2,3,4,5,6,7,8,9,10,11,12,13,14,15);
}

// ---------------- fp32 -> bf16 convert ----------------
__global__ void cvt_f32_bf16(const float* __restrict__ in, __bf16* __restrict__ out, int n) {
    int i = blockIdx.x * blockDim.x + threadIdx.x;
    if (i < n) out[i] = (__bf16)in[i];
}

// ---------------- W[out,in] -> swizzled B-panel of W^T ----------------
// B(k,n) = W[n,k]. Per (kt,nt) tile, lane l holds column n = nt*16+(l&15),
// rows k = kt*32 + (l<16?0:16) + i, i=0..15, stored contiguously (32B/lane).
__global__ void swz_w(const float* __restrict__ W, __bf16* __restrict__ Wsw, int total) {
    int idx = blockIdx.x * blockDim.x + threadIdx.x;
    if (idx >= total) return;
    int i  = idx & 15;
    int l  = (idx >> 4) & 31;
    int nt = (idx >> 9) % NT_ALL;
    int kt = idx / (512 * NT_ALL);
    int n = nt * 16 + (l & 15);
    int k = kt * 32 + ((l >> 4) << 4) + i;
    Wsw[idx] = (__bf16)W[(size_t)n * DMODEL + k];
}

// ---------------- GEMM: C[4096x768] = A[4096x768] * B + bias ----------------
// Each wave: 32 rows x 64 cols (8 C tiles); B fragments reused across the two
// row blocks -> 8 global loads per 8 WMMAs.
// MODE 0: out bf16 row-major [4096,768]   (Q, K)
// MODE 1: out bf16 transposed [768,4096]  (V -> Vt)
// MODE 2: out fp32 row-major              (final projection)
template<int MODE>
__global__ __launch_bounds__(128)
void gemm_bf16(const __bf16* __restrict__ A, const __bf16* __restrict__ Bsw,
               const float* __restrict__ bias, void* __restrict__ Out,
               float outScale) {
    int lane = threadIdx.x & 31;
    int wave = threadIdx.x >> 5;
    int m0 = blockIdx.x * 128 + wave * 32;
    int ntBase = blockIdx.y * 4;

    const __bf16* arow0 = A + (size_t)(m0 + (lane & 15)) * DMODEL;
    const __bf16* arow1 = arow0 + (size_t)16 * DMODEL;

    v8f c[2][4];
#pragma unroll
    for (int i = 0; i < 2; ++i)
#pragma unroll
        for (int j = 0; j < 4; ++j)
#pragma unroll
            for (int v = 0; v < 8; ++v) c[i][j][v] = 0.0f;

    for (int kt = 0; kt < KT_ALL; ++kt) {
        v16bf a0 = load_a_frag(arow0, kt * 32, lane);
        v16bf a1 = load_a_frag(arow1, kt * 32, lane);
#pragma unroll
        for (int j = 0; j < 4; ++j) {
            v16bf b = *(const v16bf*)(Bsw +
                ((((size_t)kt * NT_ALL) + (ntBase + j)) * 32 + lane) * 16);
            c[0][j] = wmma_bf16(a0, b, c[0][j]);
            c[1][j] = wmma_bf16(a1, b, c[1][j]);
        }
    }

    int ncol  = lane & 15;
    int mrow0 = (lane < 16) ? 0 : 8;
#pragma unroll
    for (int i = 0; i < 2; ++i)
#pragma unroll
        for (int j = 0; j < 4; ++j) {
            int n = (ntBase + j) * 16 + ncol;
            float bv = bias[n];
#pragma unroll
            for (int v = 0; v < 8; ++v) {
                int m = m0 + i * 16 + mrow0 + v;
                float val = (c[i][j][v] + bv) * outScale;
                if (MODE == 0)      ((__bf16*)Out)[(size_t)m * DMODEL + n] = (__bf16)val;
                else if (MODE == 1) ((__bf16*)Out)[(size_t)n * SEQ    + m] = (__bf16)val;
                else                ((float*)Out) [(size_t)m * DMODEL + n] = val;
            }
        }
}

// ---------------- Flash attention (transposed, LDS-free, split loops) -------
// One wave per (head, 32-query block). S^T = K*Q^T: each lane owns one query
// column per query-tile; softmax stats lane-local (+1 xor-16 combine); P^T
// B-fragment built with 8 xor-16 shuffles; O^T = V^T*P^T with per-lane rescale.
// K/V fragments are shared across both query tiles (16 loads : 16 WMMAs).
// MASK=false for interior (provably unmasked) key blocks; one masked diagonal
// step per wave since q0 is 32-aligned.
template<bool MASK>
__device__ __forceinline__ void attn_step(
    int t0, int q0, int lane, int hd,
    const __bf16* __restrict__ Km, const __bf16* __restrict__ Vt,
    const v16bf (&qb)[2][2], v8f (&o)[2][4], float (&aMax)[2], float (&aSum)[2])
{
    int  ncol   = lane & 15;
    bool hihalf = lane >= 16;
    int  mrow0  = hihalf ? 8 : 0;

    // K A-fragments for key rows t0..t0+15 and t0+16..t0+31 (shared over qt)
    const __bf16* krow1 = Km + (size_t)(t0 + ncol) * DMODEL + hd;
    const __bf16* krow2 = krow1 + (size_t)16 * DMODEL;
    v16bf ka0 = load_a_frag(krow1, 0, lane);
    v16bf ka1 = load_a_frag(krow1, 32, lane);
    v16bf kc0 = load_a_frag(krow2, 0, lane);
    v16bf kc1 = load_a_frag(krow2, 32, lane);

    // V^T A-fragments (shared over qt)
    v16bf va[4];
#pragma unroll
    for (int dt = 0; dt < 4; ++dt)
        va[dt] = load_a_frag(Vt + (size_t)(hd + dt * 16 + ncol) * SEQ, t0, lane);

#pragma unroll
    for (int qt = 0; qt < 2; ++qt) {
        v8f s1, s2;
#pragma unroll
        for (int v = 0; v < 8; ++v) { s1[v] = 0.0f; s2[v] = 0.0f; }
        s1 = wmma_bf16(ka0, qb[qt][0], s1);
        s1 = wmma_bf16(ka1, qb[qt][1], s1);
        s2 = wmma_bf16(kc0, qb[qt][0], s2);
        s2 = wmma_bf16(kc1, qb[qt][1], s2);

        float p1[8], p2[8];
        float mx = -1e30f;
        if (MASK) {
            int r  = q0 + qt * 16 + ncol;   // this lane's query row
            int tb = t0 + mrow0;
#pragma unroll
            for (int v = 0; v < 8; ++v) {
                float a = (tb + v      <= r) ? s1[v] : -1e30f;
                float b = (tb + 16 + v <= r) ? s2[v] : -1e30f;
                p1[v] = a; p2[v] = b;
                mx = fmaxf(mx, fmaxf(a, b));
            }
        } else {
#pragma unroll
            for (int v = 0; v < 8; ++v) {
                p1[v] = s1[v]; p2[v] = s2[v];
                mx = fmaxf(mx, fmaxf(s1[v], s2[v]));
            }
        }
        mx = fmaxf(mx, __shfl_xor(mx, 16));
        float mNew   = fmaxf(aMax[qt], mx);
        float rscale = __expf(aMax[qt] - mNew);
        aMax[qt] = mNew;

        float sum = 0.0f;
#pragma unroll
        for (int v = 0; v < 8; ++v) {
            p1[v] = __expf(p1[v] - mNew);
            p2[v] = __expf(p2[v] - mNew);
            sum += p1[v] + p2[v];
        }
        sum += __shfl_xor(sum, 16);
        aSum[qt] = aSum[qt] * rscale + sum;

        // P^T B-fragment (k rows = t0+ksel..+15) via xor-16 exchange
        v16bf pb;
#pragma unroll
        for (int v = 0; v < 8; ++v) {
            float send = hihalf ? p1[v] : p2[v];
            float recv = __shfl_xor(send, 16);
            pb[v]     = (__bf16)(hihalf ? recv  : p1[v]);
            pb[8 + v] = (__bf16)(hihalf ? p2[v] : recv);
        }

#pragma unroll
        for (int dt = 0; dt < 4; ++dt) {
#pragma unroll
            for (int v = 0; v < 8; ++v) o[qt][dt][v] *= rscale;
            o[qt][dt] = wmma_bf16(va[dt], pb, o[qt][dt]);
        }
    }
}

__global__ __launch_bounds__(128)
void attn_kernel(const __bf16* __restrict__ Q, const __bf16* __restrict__ Km,
                 const __bf16* __restrict__ Vt, __bf16* __restrict__ Out) {
    int lane = threadIdx.x & 31;
    int wave = threadIdx.x >> 5;
    int h  = blockIdx.y;
    int q0 = (blockIdx.x * 4 + wave) * 32;
    int hd = h * HDIM;

    int ncol  = lane & 15;
    int ksel  = (lane >= 16) ? 16 : 0;
    int mrow0 = (lane >= 16) ? 8 : 0;

    // Q as B-fragments for both query tiles
    v16bf qb[2][2];
#pragma unroll
    for (int qt = 0; qt < 2; ++qt) {
        const __bf16* qc = Q + (size_t)(q0 + qt * 16 + ncol) * DMODEL + hd + ksel;
        qb[qt][0] = *(const v16bf*)qc;
        qb[qt][1] = *(const v16bf*)(qc + 32);
    }

    v8f o[2][4];
#pragma unroll
    for (int qt = 0; qt < 2; ++qt)
#pragma unroll
        for (int dt = 0; dt < 4; ++dt)
#pragma unroll
            for (int v = 0; v < 8; ++v) o[qt][dt][v] = 0.0f;
    float aMax[2] = { -1e30f, -1e30f };
    float aSum[2] = { 0.0f, 0.0f };

    // Interior: all keys t0..t0+31 <= q0-1 < any query row -> no masking
    for (int t0 = 0; t0 < q0; t0 += 32)
        attn_step<false>(t0, q0, lane, hd, Km, Vt, qb, o, aMax, aSum);
    // Diagonal block (exactly one, q0 is 32-aligned): masked
    attn_step<true>(q0, q0, lane, hd, Km, Vt, qb, o, aMax, aSum);

    // Normalize + store: O^T(d,q) -> Out[q, hd+d]; 16B vector stores
#pragma unroll
    for (int qt = 0; qt < 2; ++qt) {
        float inv = 1.0f / aSum[qt];
#pragma unroll
        for (int dt = 0; dt < 4; ++dt) {
            v8bf ov;
#pragma unroll
            for (int v = 0; v < 8; ++v) ov[v] = (__bf16)(o[qt][dt][v] * inv);
            *(v8bf*)(Out + (size_t)(q0 + qt * 16 + ncol) * DMODEL
                         + hd + dt * 16 + mrow0) = ov;
        }
    }
}

extern "C" void kernel_launch(void* const* d_in, const int* in_sizes, int n_in,
                              void* d_out, int out_size, void* d_ws, size_t ws_size,
                              hipStream_t stream) {
    (void)in_sizes; (void)n_in; (void)out_size; (void)ws_size;
    const float* X  = (const float*)d_in[0];
    const float* Wq = (const float*)d_in[1];
    const float* bq = (const float*)d_in[2];
    const float* Wk = (const float*)d_in[3];
    const float* bk = (const float*)d_in[4];
    const float* Wv = (const float*)d_in[5];
    const float* bv = (const float*)d_in[6];
    const float* Wo = (const float*)d_in[7];
    const float* bo = (const float*)d_in[8];
    float* out = (float*)d_out;

    const size_t XE = (size_t)SEQ * DMODEL;        // 3,145,728 elements
    const int    WT = KT_ALL * NT_ALL * 32 * 16;   // 589,824 elements per W

    char* ws = (char*)d_ws;                        // ~34.5 MB total
    __bf16* Xbf  = (__bf16*)ws; ws += XE * 2;
    __bf16* WqS  = (__bf16*)ws; ws += (size_t)WT * 2;
    __bf16* WkS  = (__bf16*)ws; ws += (size_t)WT * 2;
    __bf16* WvS  = (__bf16*)ws; ws += (size_t)WT * 2;
    __bf16* WoS  = (__bf16*)ws; ws += (size_t)WT * 2;
    __bf16* Qbf  = (__bf16*)ws; ws += XE * 2;
    __bf16* Kbf  = (__bf16*)ws; ws += XE * 2;
    __bf16* Vtbf = (__bf16*)ws; ws += XE * 2;
    __bf16* Abf  = (__bf16*)ws; ws += XE * 2;

    cvt_f32_bf16<<<(int)((XE + 255) / 256), 256, 0, stream>>>(X, Xbf, (int)XE);
    swz_w<<<(WT + 255) / 256, 256, 0, stream>>>(Wq, WqS, WT);
    swz_w<<<(WT + 255) / 256, 256, 0, stream>>>(Wk, WkS, WT);
    swz_w<<<(WT + 255) / 256, 256, 0, stream>>>(Wv, WvS, WT);
    swz_w<<<(WT + 255) / 256, 256, 0, stream>>>(Wo, WoS, WT);

    dim3 gg(SEQ / 128, DMODEL / 64);               // (32,12), 4 waves/block
    const float qscale = 0.125f;                   // 1/sqrt(64) folded into Q
    gemm_bf16<0><<<gg, 128, 0, stream>>>(Xbf, WqS, bq, Qbf,  qscale);
    gemm_bf16<0><<<gg, 128, 0, stream>>>(Xbf, WkS, bk, Kbf,  1.0f);
    gemm_bf16<1><<<gg, 128, 0, stream>>>(Xbf, WvS, bv, Vtbf, 1.0f);

    attn_kernel<<<dim3(SEQ / 128, HEADS), 128, 0, stream>>>(Qbf, Kbf, Vtbf, Abf);

    gemm_bf16<2><<<gg, 128, 0, stream>>>(Abf, WoS, bo, out, 1.0f);
}
